// ARDecoderECD_4148938408893
// MI455X (gfx1250) — matrix operand order repaired and checked by reading, not compiled, not run on hardware
//
#include <hip/hip_runtime.h>

typedef _Float16 f16;
typedef __attribute__((ext_vector_type(16))) _Float16 v16h;
typedef __attribute__((ext_vector_type(8)))  _Float16 v8h;
typedef __attribute__((ext_vector_type(8)))  float    v8f;

#define TT 26
#define VV 23
#define EE 100
#define HH 128
#define LL 64
#define BB 8192

__device__ __forceinline__ v8f wmma32(v16h a, v16h b, v8f c) {
  // v_wmma_f32_16x16x32_f16: D = A(16x32) * B(32x16) + C(16x16 f32)
  return __builtin_amdgcn_wmma_f32_16x16x32_f16(false, a, false, b, (short)0, c,
                                                false, false);
}

__device__ __forceinline__ v16h frag_join(v8h lo, v8h hi) {
  v16h r;
#pragma unroll
  for (int i = 0; i < 8; ++i) { r[i] = lo[i]; r[i + 8] = hi[i]; }
  return r;
}

// A-matrix fragment (16x32 f16). lane: m = l%16, hf = l/16.
// VGPR0-3 hold k = kt*32 + 8*hf + {0..7}; VGPR4-7 hold k = kt*32 + 16 + 8*hf + {0..7}.
__device__ __forceinline__ v16h load_a_frag(const f16* row, int ld, int kt, int m, int hf) {
  const f16* p = row + m * ld + kt * 32 + 8 * hf;
  return frag_join(*(const v8h*)p, *(const v8h*)(p + 16));
}

// B-matrix fragment (32x16 f16). lane: n = l%16, hf = l/16.
// Elements 0-7 hold k = kt*32 + 8*hf + {0..7}; elements 8-15 hold k = kt*32 + 16 + 8*hf + {0..7}.
__device__ __forceinline__ v16h load_b_frag(const f16* base, int ld, int kt, int n, int hf) {
  const f16* p = base + n * ld + kt * 32 + 8 * hf;
  return frag_join(*(const v8h*)p, *(const v8h*)(p + 16));
}

// Branchless activations on v_exp_f32 / v_rcp_f32 (no EXEC-divergent libm paths).
#define LOG2E 1.4426950408889634f
__device__ __forceinline__ float fast_sigmoid(float x) {
  float e = __builtin_amdgcn_exp2f(-LOG2E * x);
  return __builtin_amdgcn_rcpf(1.0f + e);   // exp2(+inf)=inf -> rcp -> 0; exp2(-inf)=0 -> 1
}
__device__ __forceinline__ float fast_tanh(float x) {
  float e = __builtin_amdgcn_exp2f((2.0f * LOG2E) * x);
  return 1.0f - 2.0f * __builtin_amdgcn_rcpf(1.0f + e);
}

// ---------------------------------------------------------------------------
// Kernel 1: h0 = tanh(z @ fc_z_w^T + b), then GRU scan over T.
// Writes y (B,T,H) as f16 to workspace.
// ---------------------------------------------------------------------------
__global__ __launch_bounds__(128) void gru_phase(
    const float* __restrict__ z, const int* __restrict__ x_in,
    const float* __restrict__ emb, const float* __restrict__ fcw,
    const float* __restrict__ fcb, const float* __restrict__ wih,
    const float* __restrict__ whh, const float* __restrict__ bih,
    const float* __restrict__ bhh, f16* __restrict__ y_out) {
  __shared__ f16 sWih[384 * HH];     // [3H][E pad 128]
  __shared__ f16 sWhh[384 * HH];     // [3H][H]
  __shared__ f16 sEmb[24 * HH];      // [V pad 24][E pad 128]
  __shared__ f16 sFcw[HH * LL];      // [H][L]
  __shared__ float sBih[384], sBhh[384], sFcb[HH];
  __shared__ f16 sAct[2][4][16 * HH];  // per-wave ping-pong h buffers

  const int tid = threadIdx.x;
  for (int i = tid; i < 384 * HH; i += 128) {
    int n = i >> 7, k = i & 127;
    sWih[i] = (f16)(k < EE ? wih[n * EE + k] : 0.0f);
    sWhh[i] = (f16)whh[n * HH + k];
  }
  for (int i = tid; i < 24 * HH; i += 128) {
    int vv = i >> 7, k = i & 127;
    sEmb[i] = (f16)((vv < VV && k < EE) ? emb[vv * EE + k] : 0.0f);
  }
  for (int i = tid; i < HH * LL; i += 128) sFcw[i] = (f16)fcw[i];
  for (int i = tid; i < 384; i += 128) { sBih[i] = bih[i]; sBhh[i] = bhh[i]; }
  if (tid < HH) sFcb[tid] = fcb[tid];

  const int lane = tid & 31, wv = tid >> 5;
  const int ln = lane & 15, hf = lane >> 4;
  const int R = blockIdx.x * 64 + wv * 16;

  // stage this wave's z tile (16 x 64) as f16
  f16* zbuf = sAct[1][wv];
  for (int i = lane; i < 16 * LL; i += 32) {
    int m = i >> 6, k = i & 63;
    zbuf[m * LL + k] = (f16)z[(size_t)(R + m) * LL + k];
  }
  __syncthreads();

  f16* cur = sAct[0][wv];
  // h0 = tanh(z @ fcw^T + fcb)
  for (int nt = 0; nt < 8; ++nt) {
    const int n = nt * 16 + ln;
    v8f acc = {};
#pragma unroll
    for (int kt = 0; kt < 2; ++kt) {
      v16h a = load_a_frag(zbuf, LL, kt, ln, hf);
      v16h b = load_b_frag(sFcw, LL, kt, n, hf);
      acc = wmma32(a, b, acc);
    }
    const float bb = sFcb[n];
#pragma unroll
    for (int v2 = 0; v2 < 8; ++v2)
      cur[(v2 + 8 * hf) * HH + n] = (f16)fast_tanh(acc[v2] + bb);
  }
  __syncthreads();

  f16* nxt = sAct[1][wv];
  for (int t = 0; t < TT; ++t) {
    const int tok = x_in[(size_t)(R + ln) * TT + t];
    v16h ae[4], ah[4];
#pragma unroll
    for (int kt = 0; kt < 4; ++kt) {
      ae[kt] = load_a_frag(sEmb + tok * HH, HH, kt, 0, hf);
      ah[kt] = load_a_frag(cur, HH, kt, ln, hf);
    }
    for (int nt = 0; nt < 8; ++nt) {
      const int n = nt * 16 + ln;
      v8f ar = {}, az = {}, axn = {}, ahn = {};
#pragma unroll
      for (int kt = 0; kt < 4; ++kt) {
        ar  = wmma32(ae[kt], load_b_frag(sWih, HH, kt, n,       hf), ar);
        ar  = wmma32(ah[kt], load_b_frag(sWhh, HH, kt, n,       hf), ar);
        az  = wmma32(ae[kt], load_b_frag(sWih, HH, kt, n + 128, hf), az);
        az  = wmma32(ah[kt], load_b_frag(sWhh, HH, kt, n + 128, hf), az);
        axn = wmma32(ae[kt], load_b_frag(sWih, HH, kt, n + 256, hf), axn);
        ahn = wmma32(ah[kt], load_b_frag(sWhh, HH, kt, n + 256, hf), ahn);
      }
      const float br  = sBih[n] + sBhh[n];
      const float bz  = sBih[n + 128] + sBhh[n + 128];
      const float bxn = sBih[n + 256];
      const float bhn = sBhh[n + 256];
#pragma unroll
      for (int v2 = 0; v2 < 8; ++v2) {
        const int m = v2 + 8 * hf;
        float r  = fast_sigmoid(ar[v2] + br);
        float zg = fast_sigmoid(az[v2] + bz);
        float nn = fast_tanh(axn[v2] + bxn + r * (ahn[v2] + bhn));
        float hold = (float)cur[m * HH + n];
        float hnew = fmaf(zg, hold - nn, nn);   // (1-z)*n + z*h
        nxt[m * HH + n] = (f16)hnew;
        y_out[((size_t)(R + m) * TT + t) * HH + n] = (f16)hnew;
      }
    }
    f16* tmp = cur; cur = nxt; nxt = tmp;
    __syncthreads();
  }
}

// ---------------------------------------------------------------------------
// Kernel 2: LSTM scan over T (h0=c0=0) + logits = h @ out_w^T + out_b.
// Gate dim 4E=400 repacked with per-gate stride 112 (7 N-tiles each, zero pad).
// ---------------------------------------------------------------------------
__global__ __launch_bounds__(128) void lstm_phase(
    const f16* __restrict__ y_in,
    const float* __restrict__ wih, const float* __restrict__ whh,
    const float* __restrict__ bih, const float* __restrict__ bhh,
    const float* __restrict__ ow, const float* __restrict__ ob,
    float* __restrict__ out) {
  __shared__ f16 sWih[448 * HH];   // [gate*112+e][H]
  __shared__ f16 sWhh[448 * HH];   // [gate*112+e][E pad 128]
  __shared__ f16 sOw[32 * HH];     // [V pad 32][E pad 128]
  __shared__ float sBih[448], sBhh[448], sOb[32];
  __shared__ f16 sAct[2][4][16 * HH];
  __shared__ float sC[4][16 * 112];

  const int tid = threadIdx.x;
  for (int i = tid; i < 448 * HH; i += 128) {
    int n = i >> 7, k = i & 127;
    int g = n / 112, e = n - g * 112;
    float a = 0.0f, b = 0.0f;
    if (e < EE) {
      int row = g * EE + e;
      a = wih[row * HH + k];
      if (k < EE) b = whh[row * EE + k];
    }
    sWih[i] = (f16)a;
    sWhh[i] = (f16)b;
  }
  for (int i = tid; i < 32 * HH; i += 128) {
    int n = i >> 7, k = i & 127;
    sOw[i] = (f16)((n < VV && k < EE) ? ow[n * EE + k] : 0.0f);
  }
  for (int i = tid; i < 448; i += 128) {
    int g = i / 112, e = i - g * 112;
    sBih[i] = (e < EE) ? bih[g * EE + e] : 0.0f;
    sBhh[i] = (e < EE) ? bhh[g * EE + e] : 0.0f;
  }
  if (tid < 32) sOb[tid] = (tid < VV) ? ob[tid] : 0.0f;

  const int lane = tid & 31, wv = tid >> 5;
  const int ln = lane & 15, hf = lane >> 4;
  const int R = blockIdx.x * 64 + wv * 16;

  f16* cur = sAct[0][wv];
  f16* nxt = sAct[1][wv];
  float* cb = sC[wv];
  for (int i = lane; i < 16 * HH; i += 32) { cur[i] = (f16)0.0f; nxt[i] = (f16)0.0f; }
  for (int i = lane; i < 16 * 112; i += 32) cb[i] = 0.0f;
  __syncthreads();

  for (int t = 0; t < TT; ++t) {
    v16h ay[4], ah[4];
    const f16* yrow = y_in + ((size_t)(R + ln) * TT + t) * HH;
#pragma unroll
    for (int kt = 0; kt < 4; ++kt) {
      ay[kt] = load_a_frag(yrow, HH, kt, 0, hf);
      ah[kt] = load_a_frag(cur, HH, kt, ln, hf);
    }
    for (int nt = 0; nt < 7; ++nt) {
      const int n = nt * 16 + ln;
      v8f gi = {}, gf = {}, gg = {}, go = {};
#pragma unroll
      for (int kt = 0; kt < 4; ++kt) {
        gi = wmma32(ay[kt], load_b_frag(sWih, HH, kt, n,       hf), gi);
        gi = wmma32(ah[kt], load_b_frag(sWhh, HH, kt, n,       hf), gi);
        gf = wmma32(ay[kt], load_b_frag(sWih, HH, kt, n + 112, hf), gf);
        gf = wmma32(ah[kt], load_b_frag(sWhh, HH, kt, n + 112, hf), gf);
        gg = wmma32(ay[kt], load_b_frag(sWih, HH, kt, n + 224, hf), gg);
        gg = wmma32(ah[kt], load_b_frag(sWhh, HH, kt, n + 224, hf), gg);
        go = wmma32(ay[kt], load_b_frag(sWih, HH, kt, n + 336, hf), go);
        go = wmma32(ah[kt], load_b_frag(sWhh, HH, kt, n + 336, hf), go);
      }
      const float bi = sBih[n] + sBhh[n];
      const float bf = sBih[n + 112] + sBhh[n + 112];
      const float bg = sBih[n + 224] + sBhh[n + 224];
      const float bo = sBih[n + 336] + sBhh[n + 336];
#pragma unroll
      for (int v2 = 0; v2 < 8; ++v2) {
        const int m = v2 + 8 * hf;
        float iv = fast_sigmoid(gi[v2] + bi);
        float fv = fast_sigmoid(gf[v2] + bf);
        float gv = fast_tanh(gg[v2] + bg);
        float ov = fast_sigmoid(go[v2] + bo);
        float cold = cb[m * 112 + n];
        float cnew = fmaf(fv, cold, iv * gv);
        cb[m * 112 + n] = cnew;
        nxt[m * HH + n] = (f16)(ov * fast_tanh(cnew));
      }
    }
    __syncthreads();

    // logits_t = h_t @ ow^T + ob  (V=23, padded to 2 N-tiles)
    v16h a2[4];
#pragma unroll
    for (int kt = 0; kt < 4; ++kt) a2[kt] = load_a_frag(nxt, HH, kt, ln, hf);
#pragma unroll
    for (int nt = 0; nt < 2; ++nt) {
      const int n = nt * 16 + ln;
      v8f acc = {};
#pragma unroll
      for (int kt = 0; kt < 4; ++kt)
        acc = wmma32(a2[kt], load_b_frag(sOw, HH, kt, n, hf), acc);
      if (n < VV) {
        const float b2 = sOb[n];
#pragma unroll
        for (int v2 = 0; v2 < 8; ++v2) {
          const int m = v2 + 8 * hf;
          out[((size_t)(R + m) * TT + t) * VV + n] = acc[v2] + b2;
        }
      }
    }
    f16* tmp = cur; cur = nxt; nxt = tmp;
    __syncthreads();
  }
}

extern "C" void kernel_launch(void* const* d_in, const int* in_sizes, int n_in,
                              void* d_out, int out_size, void* d_ws, size_t ws_size,
                              hipStream_t stream) {
  const float* z        = (const float*)d_in[0];
  const int*   x_in     = (const int*)d_in[1];
  const float* emb      = (const float*)d_in[2];
  const float* fc_z_w   = (const float*)d_in[3];
  const float* fc_z_b   = (const float*)d_in[4];
  const float* gru_wih  = (const float*)d_in[5];
  const float* gru_whh  = (const float*)d_in[6];
  const float* gru_bih  = (const float*)d_in[7];
  const float* gru_bhh  = (const float*)d_in[8];
  const float* lstm_wih = (const float*)d_in[9];
  const float* lstm_whh = (const float*)d_in[10];
  const float* lstm_bih = (const float*)d_in[11];
  const float* lstm_bhh = (const float*)d_in[12];
  const float* out_w    = (const float*)d_in[13];
  const float* out_b    = (const float*)d_in[14];
  float* out = (float*)d_out;
  f16* ws_y = (f16*)d_ws;  // (B,T,128) f16 intermediate: 54.5 MB

  dim3 grid(BB / 64), block(128);
  gru_phase<<<grid, block, 0, stream>>>(z, x_in, emb, fc_z_w, fc_z_b,
                                        gru_wih, gru_whh, gru_bih, gru_bhh, ws_y);
  lstm_phase<<<grid, block, 0, stream>>>(ws_y, lstm_wih, lstm_whh, lstm_bih,
                                         lstm_bhh, out_w, out_b, out);
}
